// MultiheadNetwork_57131654971687
// MI455X (gfx1250) — matrix operand order repaired and checked by reading, not compiled, not run on hardware
//
#include <hip/hip_runtime.h>

// Problem dims (fixed by the reference)
#define B_ROWS   16384
#define K_IN     784
#define K_PAD    800          // 25 * 32, zero padded
#define FEAT_N   1024
#define NHEADS   64
#define OUT_D    64

typedef __attribute__((ext_vector_type(16))) __bf16       v16bf;
typedef __attribute__((ext_vector_type(8)))  float        v8f;
typedef __attribute__((ext_vector_type(4)))  unsigned int v4u;
typedef __attribute__((ext_vector_type(8)))  unsigned int v8u;

static __device__ __forceinline__ unsigned short f2bf(float f) {
  unsigned int u = __float_as_uint(f);
  u += 0x7FFFu + ((u >> 16) & 1u);     // round-to-nearest-even
  return (unsigned short)(u >> 16);
}

// A-fragment (16x32 bf16): per lane two contiguous 8-elem (16B) chunks.
static __device__ __forceinline__ v16bf load_a_frag(const unsigned short* p) {
  v4u lo = *(const v4u*)(p);
  v4u hi = *(const v4u*)(p + 8);
  v8u r;
  r[0] = lo[0]; r[1] = lo[1]; r[2] = lo[2]; r[3] = lo[3];
  r[4] = hi[0]; r[5] = hi[1]; r[6] = hi[2]; r[7] = hi[3];
  return __builtin_bit_cast(v16bf, r);
}

// B-fragment (32x16 bf16): per lane 16 contiguous K values (32B).
static __device__ __forceinline__ v16bf load_b_frag(const unsigned short* p) {
  v8u r = *(const v8u*)(p);
  return __builtin_bit_cast(v16bf, r);
}

// ---------------- conversion kernels ----------------
__global__ void mh_cvt_pad(const float* __restrict__ src, unsigned short* __restrict__ dst,
                           int rows, int kin, int kpad) {
  int i = blockIdx.x * blockDim.x + threadIdx.x;
  if (i >= rows * kpad) return;
  int r = i / kpad;
  int k = i - r * kpad;
  float v = (k < kin) ? src[r * kin + k] : 0.0f;
  dst[i] = f2bf(v);
}

__global__ void mh_cvt(const float* __restrict__ src, unsigned short* __restrict__ dst, int n) {
  int i = blockIdx.x * blockDim.x + threadIdx.x;
  if (i < n) dst[i] = f2bf(src[i]);
}

// ---------------- counting sort of rows by task ----------------
__global__ void mh_hist(const int* __restrict__ task, int* __restrict__ counts, int n) {
  int i = blockIdx.x * blockDim.x + threadIdx.x;
  if (i < n) atomicAdd(&counts[task[i]], 1);
}

__global__ void mh_scan(const int* __restrict__ counts, int* __restrict__ offs) {
  if (threadIdx.x == 0 && blockIdx.x == 0) {
    int run = 0;
    for (int i = 0; i < NHEADS; ++i) { offs[i] = run; run += counts[i]; }
  }
}

__global__ void mh_scatter(const int* __restrict__ task, const int* __restrict__ offs,
                           int* __restrict__ cursor, int* __restrict__ rowlist, int n) {
  int i = blockIdx.x * blockDim.x + threadIdx.x;
  if (i < n) {
    int t = task[i];
    int p = atomicAdd(&cursor[t], 1);
    rowlist[offs[t] + p] = i;
  }
}

// ---------------- GEMM1: h = relu(x @ W_feat^T + b_feat), bf16 in / bf16 out ----
// grid (B/32, FEAT/512), block 256 (8 waves). Wave computes 32x64 tile.
__global__ __launch_bounds__(256) void mh_gemm1_relu(
    const unsigned short* __restrict__ xb,    // [B_ROWS][K_PAD] bf16
    const unsigned short* __restrict__ wfb,   // [FEAT_N][K_PAD] bf16
    const float* __restrict__ bfeat,          // [FEAT_N] f32
    unsigned short* __restrict__ hb)          // [B_ROWS][FEAT_N] bf16
{
  const int lane = threadIdx.x & 31;
  const int wave = threadIdx.x >> 5;
  const int l15  = lane & 15;
  const int half = lane >> 4;
  const int mBase = blockIdx.x * 32;
  const int nBase = blockIdx.y * 512 + wave * 64;

  const unsigned short* pa0 = xb + (mBase + l15) * K_PAD + half * 16;
  const unsigned short* pa1 = pa0 + 16 * K_PAD;
  const unsigned short* pb[4];
#pragma unroll
  for (int j = 0; j < 4; ++j)
    pb[j] = wfb + (nBase + j * 16 + l15) * K_PAD + half * 16;

  v8f zero = {0.f, 0.f, 0.f, 0.f, 0.f, 0.f, 0.f, 0.f};
  v8f acc[2][4];
#pragma unroll
  for (int i = 0; i < 2; ++i)
#pragma unroll
    for (int j = 0; j < 4; ++j) acc[i][j] = zero;

  for (int kt = 0; kt < K_PAD / 32; ++kt) {
    const int ko = kt * 32;
    v16bf a[2], b[4];
    a[0] = load_a_frag(pa0 + ko);
    a[1] = load_a_frag(pa1 + ko);
#pragma unroll
    for (int j = 0; j < 4; ++j) b[j] = load_b_frag(pb[j] + ko);
#pragma unroll
    for (int i = 0; i < 2; ++i)
#pragma unroll
      for (int j = 0; j < 4; ++j)
        acc[i][j] = __builtin_amdgcn_wmma_f32_16x16x32_bf16(
            false, a[i], false, b[j], (short)0, acc[i][j], false, false);
  }

  // epilogue: bias + relu, store bf16 h
#pragma unroll
  for (int j = 0; j < 4; ++j) {
    const int n = nBase + j * 16 + l15;
    const float bias = bfeat[n];
#pragma unroll
    for (int i = 0; i < 2; ++i) {
#pragma unroll
      for (int r = 0; r < 8; ++r) {
        const int m = mBase + i * 16 + half * 8 + r;
        float v = acc[i][j][r] + bias;
        v = v > 0.0f ? v : 0.0f;
        hb[m * FEAT_N + n] = f2bf(v);
      }
    }
  }
}

// ---------------- GEMM2: out[rows of head] = h_rows @ W_heads[head]^T + b_heads ----
// grid (maxTiles=B/16, NHEADS), block 32 (1 wave). Wave computes 16x64 tile.
__global__ __launch_bounds__(32) void mh_gemm2(
    const unsigned short* __restrict__ hb,     // [B_ROWS][FEAT_N] bf16
    const unsigned short* __restrict__ whb,    // [NHEADS][OUT_D][FEAT_N] bf16
    const float* __restrict__ bheads,          // [NHEADS][OUT_D] f32
    const int* __restrict__ counts,
    const int* __restrict__ offs,
    const int* __restrict__ rowlist,
    float* __restrict__ out)                   // [B_ROWS][OUT_D] f32
{
  const int head = blockIdx.y;
  const int cnt  = counts[head];
  const int tile = blockIdx.x;
  if (tile * 16 >= cnt) return;               // uniform exit: EXEC stays all-1 for WMMA

  const int base = offs[head];
  const int lane = threadIdx.x & 31;
  const int l15  = lane & 15;
  const int half = lane >> 4;

  // gathered A row for this lane (clamp duplicates; stores are masked later)
  int mIdx = tile * 16 + l15;
  int mClamped = mIdx < cnt ? mIdx : (cnt - 1);
  const int rowA = rowlist[base + mClamped];
  const unsigned short* pa = hb + rowA * FEAT_N + half * 16;
  const unsigned short* pb[4];
#pragma unroll
  for (int j = 0; j < 4; ++j)
    pb[j] = whb + (head * OUT_D + j * 16 + l15) * FEAT_N + half * 16;

  v8f zero = {0.f, 0.f, 0.f, 0.f, 0.f, 0.f, 0.f, 0.f};
  v8f acc[4];
#pragma unroll
  for (int j = 0; j < 4; ++j) acc[j] = zero;

  for (int kt = 0; kt < FEAT_N / 32; ++kt) {
    const int ko = kt * 32;
    v16bf a = load_a_frag(pa + ko);
#pragma unroll
    for (int j = 0; j < 4; ++j) {
      v16bf b = load_b_frag(pb[j] + ko);
      acc[j] = __builtin_amdgcn_wmma_f32_16x16x32_bf16(
          false, a, false, b, (short)0, acc[j], false, false);
    }
  }

  // epilogue: scatter rows back to out, add head bias
#pragma unroll
  for (int r = 0; r < 8; ++r) {
    const int mi = tile * 16 + half * 8 + r;
    if (mi < cnt) {
      const int bRow = rowlist[base + mi];
#pragma unroll
      for (int j = 0; j < 4; ++j) {
        const int o = j * 16 + l15;
        out[bRow * OUT_D + o] = acc[j][r] + bheads[head * OUT_D + o];
      }
    }
  }
}

// ---------------- host launch ----------------
extern "C" void kernel_launch(void* const* d_in, const int* in_sizes, int n_in,
                              void* d_out, int out_size, void* d_ws, size_t ws_size,
                              hipStream_t stream) {
  const float* x    = (const float*)d_in[0];   // [16384,784]
  const int*   task = (const int*)  d_in[1];   // [16384]
  const float* Wf   = (const float*)d_in[2];   // [1024,784]
  const float* bf_  = (const float*)d_in[3];   // [1024]
  const float* Wh   = (const float*)d_in[4];   // [64,64,1024]
  const float* bh   = (const float*)d_in[5];   // [64,64]
  float* out = (float*)d_out;

  char* ws = (char*)d_ws;
  // workspace layout (256B aligned)
  const size_t XB_OFF   = 0;                                   // 16384*800*2  = 26,214,400
  const size_t WFB_OFF  = XB_OFF  + (size_t)B_ROWS * K_PAD * 2;     // + 1,638,400
  const size_t WHB_OFF  = WFB_OFF + (size_t)FEAT_N * K_PAD * 2;     // + 8,388,608
  const size_t HB_OFF   = WHB_OFF + (size_t)NHEADS * OUT_D * FEAT_N * 2; // +33,554,432
  const size_t CNT_OFF  = HB_OFF  + (size_t)B_ROWS * FEAT_N * 2;
  const size_t CUR_OFF  = CNT_OFF + 256;
  const size_t OFFS_OFF = CUR_OFF + 256;
  const size_t LIST_OFF = OFFS_OFF + 256;

  unsigned short* xb  = (unsigned short*)(ws + XB_OFF);
  unsigned short* wfb = (unsigned short*)(ws + WFB_OFF);
  unsigned short* whb = (unsigned short*)(ws + WHB_OFF);
  unsigned short* hb  = (unsigned short*)(ws + HB_OFF);
  int* counts  = (int*)(ws + CNT_OFF);
  int* cursor  = (int*)(ws + CUR_OFF);
  int* offs    = (int*)(ws + OFFS_OFF);
  int* rowlist = (int*)(ws + LIST_OFF);

  // zero histogram counters (counts + cursor are adjacent: 512 bytes)
  hipMemsetAsync(counts, 0, 512, stream);

  // convert inputs to bf16 (x/W_feat K-padded 784 -> 800)
  {
    int n = B_ROWS * K_PAD;
    mh_cvt_pad<<<(n + 255) / 256, 256, 0, stream>>>(x, xb, B_ROWS, K_IN, K_PAD);
  }
  {
    int n = FEAT_N * K_PAD;
    mh_cvt_pad<<<(n + 255) / 256, 256, 0, stream>>>(Wf, wfb, FEAT_N, K_IN, K_PAD);
  }
  {
    int n = NHEADS * OUT_D * FEAT_N;
    mh_cvt<<<(n + 255) / 256, 256, 0, stream>>>(Wh, whb, n);
  }

  // counting sort of rows by task
  mh_hist<<<(B_ROWS + 255) / 256, 256, 0, stream>>>(task, counts, B_ROWS);
  mh_scan<<<1, 64, 0, stream>>>(counts, offs);
  mh_scatter<<<(B_ROWS + 255) / 256, 256, 0, stream>>>(task, offs, cursor, rowlist, B_ROWS);

  // GEMM1: h = relu(x @ Wf^T + bf)
  {
    dim3 grid(B_ROWS / 32, FEAT_N / 512);   // (512, 2)
    mh_gemm1_relu<<<grid, 256, 0, stream>>>(xb, wfb, bf_, hb);
  }

  // GEMM2: per-head tiles over sorted rows
  {
    dim3 grid(B_ROWS / 16, NHEADS);         // (1024, 64); most blocks early-exit
    mh_gemm2<<<grid, 32, 0, stream>>>(hb, whb, bh, counts, offs, rowlist, out);
  }
}
